// cLSTM_91182155694691
// MI455X (gfx1250) — matrix operand order repaired and checked by reading, not compiled, not run on hardware
//
#include <hip/hip_runtime.h>
#include <hip/hip_bf16.h>

typedef __attribute__((ext_vector_type(16))) _Float16 v16h;
typedef __attribute__((ext_vector_type(8)))  _Float16 v8h;
typedef __attribute__((ext_vector_type(2)))  _Float16 v2h;
typedef __attribute__((ext_vector_type(8)))  float    v8f;
typedef __attribute__((ext_vector_type(2)))  float    v2f;

#define Bsz  32
#define Tlen 512
#define Nser 32
#define Hid  128
#define G4   512   // 4*H
#define Ktot 160   // H + P (fused [h | x] K dimension)

// gfx1250 hardware tanh (TRANS op); sigmoid derived from it.
__device__ __forceinline__ float fast_tanh(float x) {
#if __has_builtin(__builtin_amdgcn_tanhf)
    return __builtin_amdgcn_tanhf(x);
#elif __has_builtin(__builtin_amdgcn_tanh_f32)
    return __builtin_amdgcn_tanh_f32(x);
#else
    return tanhf(x);
#endif
}
__device__ __forceinline__ float fast_sig(float x) {
    return 0.5f + 0.5f * fast_tanh(0.5f * x);
}

// One block per series n. 512 threads = 16 wave32 waves.
// Wave w: h-tile u = w>>1 (16 hidden columns), batch tile mi = w&1.
// Weights [W_hh | W_ih] live in VGPRs as 20 v16h WMMA B-fragments for all
// 512 steps; i/f/g/o accumulators for the same (b,h) share register slots,
// so the cell update is fully in-register.
__launch_bounds__(512, 1)
__global__ void clstm_wmma_kernel(const float* __restrict__ X,
                                  const float* __restrict__ W_ih,
                                  const float* __restrict__ W_hh,
                                  const float* __restrict__ b_ih,
                                  const float* __restrict__ b_hh,
                                  const float* __restrict__ W_out,
                                  const float* __restrict__ b_out,
                                  float* __restrict__ out)
{
    const int n    = blockIdx.x;
    const int tid  = threadIdx.x;
    const int wave = tid >> 5;
    const int lane = tid & 31;

    const int u  = wave >> 1;          // h-tile 0..7
    const int mi = wave & 1;           // batch tile 0..1

    __shared__ _Float16 h_lds[Bsz * Ktot];   // [b][k]: k<128 h, k>=128 x_t
    __shared__ float    partial[Bsz * 16];   // head partial sums

    for (int i = tid; i < Bsz * Ktot; i += 512)
        h_lds[i] = (_Float16)0.0f;

    // ---- per-thread constants hoisted out of the t-loop ----
    const int hcol = u * 16 + (lane & 15);
    float bg4[4];                            // bias per gate type, this column
    #pragma unroll
    for (int ty = 0; ty < 4; ++ty)
        bg4[ty] = b_ih[n * G4 + ty * Hid + hcol] + b_hh[n * G4 + ty * Hid + hcol];

    const int hseg = tid & 15;               // head: 8-wide column segment
    const int hb   = tid >> 4;               // head: batch row
    float wr[8];
    #pragma unroll
    for (int e = 0; e < 8; ++e) wr[e] = W_out[n * Hid + hseg * 8 + e];
    const float boutn = b_out[n];

    // X staging: thread owns float2 X[xb, :, xp..xp+1], software-pipelined.
    const int xb = tid >> 4, xp = (tid & 15) * 2;
    const float* Xptr = X + (size_t)xb * Tlen * Nser + xp;

    // ---- preload register-resident B fragments Bf[gate_type][k_chunk] ----
    v16h Bf[4][5];
    {
        const int col   = lane & 15;
        const int krow0 = (lane >> 4) * 8;
        for (int ty = 0; ty < 4; ++ty) {
            const int gate = ty * Hid + u * 16 + col;
            for (int kc = 0; kc < 5; ++kc) {
                v16h bf;
                #pragma unroll
                for (int e = 0; e < 16; ++e) {
                    const int k = kc * 32 + krow0 + (e & 7) + ((e >> 3) << 4);
                    float w;
                    if (k < Hid)
                        w = W_hh[(size_t)n * G4 * Hid + (size_t)gate * Hid + k];
                    else
                        w = W_ih[(size_t)n * G4 * Nser + (size_t)gate * Nser + (k - Hid)];
                    bf[e] = (_Float16)w;
                }
                Bf[ty][kc] = bf;
            }
        }
    }

    // Cell state: lane owns rows b = mi*16 + 8*(lane>>4) + r, column hcol.
    float creg[8];
    #pragma unroll
    for (int r = 0; r < 8; ++r) creg[r] = 0.0f;

    // A fragment (16x32 f16): row = mi*16 + (lane&15); lanes 0-15 take
    // K {0..7,16..23}, lanes 16-31 take K {8..15,24..31} of each chunk.
    const int arow  = mi * 16 + (lane & 15);
    const int akoff = (lane >> 4) * 8;
    const _Float16* abase = &h_lds[arow * Ktot + akoff];
    const int brow = mi * 16 + 8 * (lane >> 4);

    v2f xv = *(const v2f*)Xptr;              // prefetch x_0
    __syncthreads();

    #pragma unroll 1
    for (int t = 0; t < Tlen; ++t) {
        // ---- commit prefetched x_t, issue prefetch of x_{t+1} ----
        {
            v2h xh; xh[0] = (_Float16)xv[0]; xh[1] = (_Float16)xv[1];
            *(v2h*)(&h_lds[xb * Ktot + Hid + xp]) = xh;
            const int tn = (t + 1 < Tlen) ? (t + 1) : t;
            xv = *(const v2f*)(Xptr + (size_t)tn * Nser);
        }
        __syncthreads();                      // (A) x_t + prev h visible

        // ---- GEMM: per-kc A load (2x ds_load_b128) -> 4 WMMAs ----
        // acc zero-init: first WMMA of each chain folds C to inline 0.
        v8f acc[4];
        #pragma unroll
        for (int ty = 0; ty < 4; ++ty)
            #pragma unroll
            for (int e = 0; e < 8; ++e) acc[ty][e] = 0.0f;

        #pragma unroll
        for (int kc = 0; kc < 5; ++kc) {
            const v8h a0 = *(const v8h*)(abase + kc * 32);
            const v8h a1 = *(const v8h*)(abase + kc * 32 + 16);
            v16h a;
            #pragma unroll
            for (int e = 0; e < 8; ++e) { a[e] = a0[e]; a[8 + e] = a1[e]; }
            #pragma unroll
            for (int ty = 0; ty < 4; ++ty)
                acc[ty] = __builtin_amdgcn_wmma_f32_16x16x32_f16(
                    false, a, false, Bf[ty][kc], (short)0, acc[ty], false, false);
        }
        __syncthreads();                      // (B) A reads done before h rewrite

        // ---- in-register LSTM cell update (hardware tanh) ----
        #pragma unroll
        for (int r = 0; r < 8; ++r) {
            const float si = fast_sig(acc[0][r] + bg4[0]);
            const float sf = fast_sig(acc[1][r] + bg4[1]);
            const float tg = fast_tanh(acc[2][r] + bg4[2]);
            const float so = fast_sig(acc[3][r] + bg4[3]);
            const float c  = sf * creg[r] + si * tg;
            creg[r] = c;
            const float hv = so * fast_tanh(c);
            h_lds[(brow + r) * Ktot + hcol] = (_Float16)hv;
        }
        __syncthreads();                      // (C) h visible to head readers

        // ---- 1x1 conv head: one ds_load_b128 of 8 contiguous f16 ----
        {
            const v8h hvv = *(const v8h*)(&h_lds[hb * Ktot + hseg * 8]);
            float s = 0.0f;
            #pragma unroll
            for (int e = 0; e < 8; ++e) s += (float)hvv[e] * wr[e];
            partial[hb * 16 + hseg] = s;
        }
        __syncthreads();                      // (D) partials ready
        if (tid < Bsz) {
            const v8f p0 = *(const v8f*)(&partial[tid * 16]);
            const v8f p1 = *(const v8f*)(&partial[tid * 16 + 8]);
            float s = 0.0f;
            #pragma unroll
            for (int e = 0; e < 8; ++e) s += p0[e] + p1[e];
            out[(size_t)tid * Tlen * Nser + (size_t)t * Nser + n] = s + boutn;
        }
        // no end barrier needed: next writes to partial/h/x are each preceded
        // by at least one of barriers (A)(B)(C) in the next iteration.
    }

    // ---- final states hT, cT (written once, outside the hot loop) ----
    const size_t offH = (size_t)Bsz * Tlen * Nser;
    const size_t offC = offH + (size_t)Nser * Bsz * Hid;
    #pragma unroll
    for (int r = 0; r < 8; ++r) {
        const int b = brow + r;
        out[offH + (size_t)n * Bsz * Hid + (size_t)b * Hid + hcol] =
            (float)h_lds[b * Ktot + hcol];
        out[offC + (size_t)n * Bsz * Hid + (size_t)b * Hid + hcol] = creg[r];
    }
}

extern "C" void kernel_launch(void* const* d_in, const int* in_sizes, int n_in,
                              void* d_out, int out_size, void* d_ws, size_t ws_size,
                              hipStream_t stream) {
    const float* X    = (const float*)d_in[0];
    const float* W_ih = (const float*)d_in[1];
    const float* W_hh = (const float*)d_in[2];
    const float* b_ih = (const float*)d_in[3];
    const float* b_hh = (const float*)d_in[4];
    const float* W_out= (const float*)d_in[5];
    const float* b_out= (const float*)d_in[6];
    float* out = (float*)d_out;
    clstm_wmma_kernel<<<Nser, 512, 0, stream>>>(X, W_ih, W_hh, b_ih, b_hh,
                                                W_out, b_out, out);
}
